// LinearAttention_4440996184658
// MI455X (gfx1250) — compile-verified, matrix operand order checked by reading, not run on hardware
//
#include <hip/hip_runtime.h>

// ---------------------------------------------------------------------------
// Causal linear attention (ELU+1 feature map), chunked formulation for CDNA5.
//   B=2, T=2048, H=16, D=64, fp32 in/out.
//   - chunk C=32 == WMMA K  ->  v_wmma_f32_16x16x32_bf16 for all matmuls
//   - fp32 accumulation; running state S lives in WMMA accumulator VGPRs
//   - TDM double-buffered: tensor_load_to_lds for q/k/v chunks,
//     tensor_store_from_lds for output chunks (all global traffic via TDM)
//   - WMMA operands stored lane-contiguous in LDS -> ds_load_b128 fragments
//   - normalizer via v_rcp_f32 once per row, multiply per element
// ---------------------------------------------------------------------------

#define B_  2
#define T_  2048
#define H_  16
#define D_  64
#define C_  32
#define EPS_ 1e-6f

#define QS_  (D_ + 4)   // f32 stride, phi(q)/phi(k) c-major rows (272B, 16B-aligned)
#define KTS_ (C_ + 4)   // f32 stride, phi(k) d-major rows (144B)
#define PS_  (C_ + 4)   // f32 stride, masked scores rows (144B)
#define VTS_ (C_ + 40)  // bf16 stride, v   n-major rows: 72 el = 144B
#define STS_ (D_ + 8)   // bf16 stride, S mirror n-major rows: 72 el = 144B

typedef __bf16 bf16_t;
typedef bf16_t v16bf __attribute__((ext_vector_type(16)));
typedef bf16_t v8bf  __attribute__((ext_vector_type(8)));
typedef float  v8f   __attribute__((ext_vector_type(8)));
typedef float  v4f   __attribute__((ext_vector_type(4)));
typedef unsigned int v4u __attribute__((ext_vector_type(4)));
typedef unsigned int v8u __attribute__((ext_vector_type(8)));

__device__ __forceinline__ float phi_elu1(float x) {
    return x > 0.f ? x + 1.f : __expf(x);   // elu(x)+1
}

__device__ __forceinline__ v8f wmma_bf16(v16bf a, v16bf b, v8f c) {
    return __builtin_amdgcn_wmma_f32_16x16x32_bf16(
        false, a, false, b, (short)0, c, false, false);
}

// ---- Tensor Data Mover: 2D tile (32 rows x 64 f32, row stride 1024 f32) ----
// D# packing per CDNA5 ISA:
//   g0: count=1 | lds_addr | global_addr[31:0] | addr[56:32] + type=2<<30
//   g1: data_size=4B | tensor_dim0=64 | tensor_dim1=32 | tile_dim0=64 |
//       tile_dim1=32 | tensor_dim0_stride=1024
__device__ __forceinline__ void tdm_desc(const void* gaddr, unsigned int lds_byte,
                                         v4u& g0, v8u& g1) {
    unsigned long long ga = (unsigned long long)(uintptr_t)gaddr;
    g0 = (v4u){ 1u, lds_byte, (unsigned int)ga,
                ((unsigned int)(ga >> 32) & 0x01FFFFFFu) | 0x80000000u };
    g1 = (v8u){ 0x00020000u,   // data_size=2 (4 bytes); no mask/flags
                0x00400000u,   // tensor_dim0 low16 = 64  @ bits[63:48]
                0x00200000u,   // tensor_dim1 low16 = 32  @ bits[95:80]
                0x00400000u,   // tile_dim0        = 64  @ bits[127:112]
                32u,           // tile_dim1 = 32, tile_dim2 = 0
                1024u,         // tensor_dim0_stride low32 (elements)
                0u, 0u };      // dim1 stride unused (2D tile)
}

__device__ __forceinline__ void tdm_load(const float* gsrc, unsigned int lds_byte) {
    v4u g0; v8u g1; tdm_desc(gsrc, lds_byte, g0, g1);
    v4u gz = { 0u, 0u, 0u, 0u };
    asm volatile("tensor_load_to_lds %0, %1, %2, %3"
                 :: "s"(g0), "s"(g1), "s"(gz), "s"(gz) : "memory");
}

__device__ __forceinline__ void tdm_store(float* gdst, unsigned int lds_byte) {
    v4u g0; v8u g1; tdm_desc(gdst, lds_byte, g0, g1);
    v4u gz = { 0u, 0u, 0u, 0u };
    asm volatile("tensor_store_from_lds %0, %1, %2, %3"
                 :: "s"(g0), "s"(g1), "s"(gz), "s"(gz) : "memory");
}

__device__ __forceinline__ unsigned int lds_addr_of(const void* p) {
    // flat pointers in the LDS aperture carry the LDS byte offset in [31:0]
    return (unsigned int)(uintptr_t)p;
}

// ---- fragment loaders --------------------------------------------------
// 16-bit operand layout: lane half -> kb in {0,8}; element runs k=kb..kb+7
// and k=kb+16..kb+23 are contiguous -> 16B vector loads.

// row-major f32 source, 16 rows x 32 k:  frag[m][k] = src[m*stride + k]
__device__ __forceinline__ v16bf frag_f32(const float* src, int stride, int lane) {
    int m = lane & 15, kb = (lane >> 4) * 8;
    const float* r = src + m * stride + kb;
    v4f a0 = *(const v4f*)(r);
    v4f a1 = *(const v4f*)(r + 4);
    v4f a2 = *(const v4f*)(r + 16);
    v4f a3 = *(const v4f*)(r + 20);
    v16bf f;
#pragma unroll
    for (int e = 0; e < 4; ++e) {
        f[e]      = (bf16_t)a0[e];
        f[4 + e]  = (bf16_t)a1[e];
        f[8 + e]  = (bf16_t)a2[e];
        f[12 + e] = (bf16_t)a3[e];
    }
    return f;
}

// n-major bf16 source: frag[n][k] = srcT[n*stride + k]  (B operands)
__device__ __forceinline__ v16bf frag_bfT(const bf16_t* srcT, int stride, int lane) {
    int n = lane & 15, kb = (lane >> 4) * 8;
    const bf16_t* r = srcT + n * stride + kb;
    v8bf lo = *(const v8bf*)(r);
    v8bf hi = *(const v8bf*)(r + 16);
    return __builtin_shufflevector(lo, hi, 0, 1, 2, 3, 4, 5, 6, 7,
                                           8, 9, 10, 11, 12, 13, 14, 15);
}

__global__ void __launch_bounds__(128)
lin_attn_chunk_kernel(const float* __restrict__ q,
                      const float* __restrict__ k,
                      const float* __restrict__ v,
                      float* __restrict__ out) {
    // TDM landing / departure tiles (double buffered), row-major (c,d)
    __shared__ __align__(16) float rawq[2][C_ * D_];
    __shared__ __align__(16) float rawk[2][C_ * D_];
    __shared__ __align__(16) float rawv[2][C_ * D_];
    __shared__ __align__(16) float outb[2][C_ * D_];
    // operand buffers
    __shared__ __align__(16) float  qphi [C_ * QS_];   // phi(q), c-major
    __shared__ __align__(16) float  kphi [C_ * QS_];   // phi(k), c-major (scores B)
    __shared__ __align__(16) float  kphiT[D_ * KTS_];  // phi(k), d-major (state A, z)
    __shared__ __align__(16) bf16_t vT   [D_ * VTS_];  // v, n-major (B operand)
    __shared__ __align__(16) float  pmat [C_ * PS_];   // masked scores
    __shared__ __align__(16) bf16_t smatT[D_ * STS_];  // state mirror, n-major
    __shared__ float zvec[D_];
    __shared__ float dinv[C_];                         // reciprocal denominators

    const int tid  = threadIdx.x;
    const int lane = tid & 31;
    const int wave = tid >> 5;          // 0..3, also the 16-col e-slab index
    const int we   = wave;
    const int bh   = blockIdx.x;
    const int b    = bh / H_;
    const int h    = bh % H_;

    const size_t row_stride = (size_t)H_ * D_;                      // 1024
    const size_t base       = (size_t)b * T_ * row_stride + (size_t)h * D_;
    const float* qh = q + base;
    const float* kh = k + base;
    const float* vh = v + base;
    float*       oh = out + base;

    // ---- init state + kick off TDM for chunk 0 ----
    for (int i = tid; i < D_ * STS_; i += 128) smatT[i] = (bf16_t)0.f;
    if (tid < D_) zvec[tid] = 0.f;
    const v8f zero8 = {0.f, 0.f, 0.f, 0.f, 0.f, 0.f, 0.f, 0.f};
    v8f s_acc[4] = {zero8, zero8, zero8, zero8};    // fp32 state slab 64x16

    if (wave == 0) {
        tdm_load(qh, lds_addr_of(&rawq[0][0]));
        tdm_load(kh, lds_addr_of(&rawk[0][0]));
        tdm_load(vh, lds_addr_of(&rawv[0][0]));
    }
    __syncthreads();

    for (int t0 = 0; t0 < T_; t0 += C_) {
        const int cur = (t0 / C_) & 1;
        const int nxt = cur ^ 1;

        // ---- drain TDM (this chunk's loads + previous chunk's store) ----
        if (wave == 0) {
            __builtin_amdgcn_s_wait_tensorcnt(0);
        }
        __syncthreads();                    // raw[cur] visible; outb[cur] free
        if (wave == 0) {
            if (t0 + C_ < T_) {             // prefetch next chunk
                size_t off = (size_t)(t0 + C_) * row_stride;
                tdm_load(qh + off, lds_addr_of(&rawq[nxt][0]));
                tdm_load(kh + off, lds_addr_of(&rawk[nxt][0]));
                tdm_load(vh + off, lds_addr_of(&rawv[nxt][0]));
            }
            if (t0 > 0) {                   // write back previous chunk output
                tdm_store(oh + (size_t)(t0 - C_) * row_stride,
                          lds_addr_of(&outb[nxt][0]));
            }
        }

        // ---- convert raw -> phi / bf16 operand layouts ----
        for (int i = tid; i < C_ * D_; i += 128) {
            int c = i >> 6, d = i & 63;
            float pq = phi_elu1(rawq[cur][i]);
            float pk = phi_elu1(rawk[cur][i]);
            qphi [c * QS_  + d] = pq;
            kphi [c * QS_  + d] = pk;
            kphiT[d * KTS_ + c] = pk;
            vT   [d * VTS_ + c] = (bf16_t)rawv[cur][i];
        }
        __syncthreads();

        // ---- hoist reusable fragments into registers ----
        v16bf qA[2][2];
#pragma unroll
        for (int mi = 0; mi < 2; ++mi)
#pragma unroll
            for (int ks = 0; ks < 2; ++ks)
                qA[mi][ks] = frag_f32(&qphi[(mi * 16) * QS_ + ks * 32], QS_, lane);
        v16bf vB = frag_bfT(&vT[(we * 16) * VTS_], VTS_, lane);
        v16bf sB[2];
#pragma unroll
        for (int ks = 0; ks < 2; ++ks)      // previous-chunk state, own columns
            sB[ks] = frag_bfT(&smatT[(we * 16) * STS_ + ks * 32], STS_, lane);

        // ---- scores P = phiQ * phiK^T, one 16x16 tile per wave, causal mask
        {
            int mi = wave >> 1, nj = wave & 1;
            v8f p = zero8;
#pragma unroll
            for (int ks = 0; ks < 2; ++ks) {
                // B[k=d][n=c'] = kphi[c'][d]: n-major runs are contiguous
                v16bf bb = frag_f32(&kphi[(nj * 16) * QS_ + ks * 32], QS_, lane);
                p = wmma_bf16(qA[mi][ks], bb, p);
            }
            int half = lane >> 4, col = lane & 15;
#pragma unroll
            for (int r = 0; r < 8; ++r) {
                int row = mi * 16 + r + 8 * half;   // C/D layout row
                int cg  = nj * 16 + col;
                pmat[row * PS_ + cg] = (cg <= row) ? p[r] : 0.f;
            }
        }
        __syncthreads();

        // ---- reciprocal denominator: 1/(rowsum(P) + phiQ.z_prev + eps) ----
        if (wave == 0) {
            int i = lane;
            float rs = 0.f;
            for (int j = 0; j < C_; ++j) rs += pmat[i * PS_ + j];
            float dz = 0.f;
            for (int d = 0; d < D_; ++d) dz += qphi[i * QS_ + d] * zvec[d];
            dinv[i] = __builtin_amdgcn_rcpf(rs + dz + EPS_);  // den > 0
        }
        __syncthreads();

        // ---- z update (waves 1,2); den already consumed z_prev ----
        if (wave == 1 || wave == 2) {
            int d = tid - 32;               // 0..63
            float s = 0.f;
            for (int c = 0; c < C_; ++c) s += kphiT[d * KTS_ + c];
            zvec[d] += s;
        }

        // ---- output slab: O = phiQ * S_prev + P * V, into LDS out tile ----
#pragma unroll
        for (int mi = 0; mi < 2; ++mi) {
            v8f o = zero8;
#pragma unroll
            for (int ks = 0; ks < 2; ++ks)              // inter-chunk, K over D
                o = wmma_bf16(qA[mi][ks], sB[ks], o);
            {                                           // intra-chunk, K = C
                v16bf pa = frag_f32(&pmat[(mi * 16) * PS_], PS_, lane);
                o = wmma_bf16(pa, vB, o);
            }
            int half = lane >> 4, col = lane & 15;
#pragma unroll
            for (int r = 0; r < 8; ++r) {
                int cl = mi * 16 + r + 8 * half;        // row within chunk
                outb[cur][cl * D_ + we * 16 + col] = o[r] * dinv[cl];
            }
        }

        // ---- state update: S_slab += phiK^T * V_slab (fp32 accumulators) ----
#pragma unroll
        for (int md = 0; md < 4; ++md) {
            // A[m=d][k=c] = kphiT[d][c]: rows contiguous in c
            v16bf a = frag_f32(&kphiT[(md * 16) * KTS_], KTS_, lane);
            s_acc[md] = wmma_bf16(a, vB, s_acc[md]);
        }
        // refresh bf16 mirror of our own 16 state rows (n-major, wave-private)
        {
            int half = lane >> 4, col = lane & 15;
            int e = we * 16 + col;
#pragma unroll
            for (int md = 0; md < 4; ++md) {
                v8bf pk;
#pragma unroll
                for (int r = 0; r < 8; ++r) pk[r] = (bf16_t)s_acc[md][r];
                *(v8bf*)&smatT[e * STS_ + md * 16 + 8 * half] = pk;
            }
        }
        __syncthreads();    // outb[cur] complete; protect buffers for next iter
    }

    // ---- write back the final chunk (implicit wait-idle before ENDPGM) ----
    if (wave == 0) {
        const int lastbuf = ((T_ / C_) - 1) & 1;
        tdm_store(oh + (size_t)(T_ - C_) * row_stride,
                  lds_addr_of(&outb[lastbuf][0]));
    }
}

extern "C" void kernel_launch(void* const* d_in, const int* in_sizes, int n_in,
                              void* d_out, int out_size, void* d_ws, size_t ws_size,
                              hipStream_t stream) {
    (void)in_sizes; (void)n_in; (void)out_size; (void)d_ws; (void)ws_size;
    const float* q = (const float*)d_in[0];
    const float* k = (const float*)d_in[1];
    const float* v = (const float*)d_in[2];
    float* out     = (float*)d_out;

    dim3 grid(B_ * H_);   // one workgroup per (b,h) head-sequence
    dim3 block(128);      // 4 wave32 waves; each owns a 16-col state slab
    lin_attn_chunk_kernel<<<grid, block, 0, stream>>>(q, k, v, out);
}
